// AttentionAggregator2_66125316489282
// MI455X (gfx1250) — matrix operand order repaired and checked by reading, not compiled, not run on hardware
//
#include <hip/hip_runtime.h>
#include <hip/hip_bf16.h>
#include <math.h>

// ---------------------------------------------------------------------------
// Types for WMMA fragments (wave32, gfx1250)
// ---------------------------------------------------------------------------
typedef __attribute__((ext_vector_type(16))) _Float16 v16h;
typedef __attribute__((ext_vector_type(8)))  _Float16 v8h;
typedef __attribute__((ext_vector_type(8)))  float    v8f;
typedef __attribute__((ext_vector_type(4)))  float    v4f;
typedef __attribute__((ext_vector_type(4)))  int      v4i;

#define ACT_NONE      0
#define ACT_TANH      1
#define ACT_BIAS_RELU 2

// ---------------------------------------------------------------------------
// Async global->LDS staging (CDNA5). Falls back to load+ds_store if the
// builtin is unavailable. Probe-verified signature on this toolchain:
// (v4i AS(1)*, v4i AS(3)*, imm offset, imm cpol).
// ---------------------------------------------------------------------------
#if defined(__has_builtin)
#if __has_builtin(__builtin_amdgcn_global_load_async_to_lds_b128)
#define USE_ASYNC_LDS 1
#endif
#endif

#if defined(USE_ASYNC_LDS)
typedef v4i __attribute__((address_space(1))) gas_v4i;
typedef v4i __attribute__((address_space(3))) las_v4i;
__device__ __forceinline__ void stage16B(const _Float16* g, _Float16* l) {
    __builtin_amdgcn_global_load_async_to_lds_b128(
        (gas_v4i*)(uintptr_t)g,
        (las_v4i*)(unsigned int)(uintptr_t)l, 0, 0);
}
#if __has_builtin(__builtin_amdgcn_s_wait_asynccnt)
#define WAIT_ASYNC(n) __builtin_amdgcn_s_wait_asynccnt(n)
#else
#define WAIT_ASYNC(n) asm volatile("s_wait_asynccnt %0" :: "i"(n) : "memory")
#endif
#else
__device__ __forceinline__ void stage16B(const _Float16* g, _Float16* l) {
    *(v8h*)l = *(const v8h*)g;      // global_load_b128 + ds_store_b128
}
#define WAIT_ASYNC(n) do {} while (0)
#endif

// Stage one 32(K) x 128(N) f16 B tile into LDS: 8KB = 512 x 16B chunks,
// 2 chunks per thread (256 threads). LDS layout: [col 0..127][32 K halves].
__device__ __forceinline__ void stageTile(const _Float16* __restrict__ W, int n0,
                                          int ldw, int kb, _Float16* dst, int tid) {
#pragma unroll
    for (int s = 0; s < 2; ++s) {
        const int c = tid + s * 256;                              // 0..511
        const _Float16* g = W + (size_t)(n0 + (c >> 2)) * ldw + kb + (c & 3) * 8;
        stage16B(g, dst + c * 8);
    }
}

// A-matrix fragment (16x32 f16, ISA layout): lane holds row (lane&15);
// halves [0..7]  = K = kb + (lane>>4)*8 .. +7
// halves [8..15] = K = kb + 16 + (lane>>4)*8 .. +7
__device__ __forceinline__ v16h loadAfrag(const _Float16* rowBaseAtKb, int khalf) {
    v8h a0 = *(const v8h*)(rowBaseAtKb + khalf * 8);
    v8h a1 = *(const v8h*)(rowBaseAtKb + 16 + khalf * 8);
    v16h r;
#pragma unroll
    for (int i = 0; i < 8; ++i) { r[i] = a0[i]; r[8 + i] = a1[i]; }
    return r;
}

// ---------------------------------------------------------------------------
// GEMM (all-f16 operands): Out[m, col_off+n] = act( sum_k A[m,k]*W[n,k] (+bias) )
// Workgroup tile 128(M) x 128(N): 8 waves, each 16 rows; B tile shared via
// double-buffered LDS with async copies; A double-buffered in registers.
// Requires M%128==0, N%128==0, K%32==0 (true for every call below).
// ---------------------------------------------------------------------------
template <typename OT, int ACT>
__global__ __launch_bounds__(256) void gemm_wmma(
    const _Float16* __restrict__ A, int lda,
    const _Float16* __restrict__ W, int ldw,
    OT* __restrict__ Out, int ldo, int col_off,
    const float* __restrict__ bias,
    int M, int N, int K)
{
    __shared__ __align__(16) _Float16 Bsh[2][128 * 32];   // 2 x 8KB

    const int tid  = threadIdx.x;
    const int lane = tid & 31;
    const int wave = tid >> 5;
    const int ntn  = N >> 7;
    const int tm = blockIdx.x / ntn, tn = blockIdx.x % ntn;
    const int m0 = tm << 7, n0 = tn << 7;
    const int row = lane & 15, khalf = lane >> 4;

    const _Float16* aRow = A + (size_t)(m0 + wave * 16 + row) * lda;

    stageTile(W, n0, ldw, 0, &Bsh[0][0], tid);            // prologue: B buf 0
    v16h a = loadAfrag(aRow, khalf);                      // prologue: A kb=0

    v8f acc[8] = {v8f{}, v8f{}, v8f{}, v8f{}, v8f{}, v8f{}, v8f{}, v8f{}};
    int buf = 0;
    for (int kb = 0; kb < K; kb += 32) {
        const bool hasNext = (kb + 32) < K;
        if (hasNext) {
            stageTile(W, n0, ldw, kb + 32, &Bsh[buf ^ 1][0], tid);
            WAIT_ASYNC(2);        // 2 outstanding (next tile); current tile landed
        } else {
            WAIT_ASYNC(0);
        }
        __builtin_prefetch(aRow + kb + 64, 0, 1);         // stream A two steps ahead
        __syncthreads();          // current B buffer visible to all waves

        // Read ALL B fragments first (16 ds_load_b128 as one clause), so the
        // dscnt waits are progressive and overlap with the WMMAs below.
        const _Float16* bcur = &Bsh[buf][0];
        v16h bf[8];
#pragma unroll
        for (int j = 0; j < 8; ++j)
            bf[j] = __builtin_shufflevector(*(const v8h*)(bcur + (j * 16 + row) * 32 + khalf * 8), *(const v8h*)(bcur + (j * 16 + row) * 32 + 16 + khalf * 8), 0, 1, 2, 3, 4, 5, 6, 7, 8, 9, 10, 11, 12, 13, 14, 15);

        // Issue next iteration's A loads before the WMMAs (register pipeline).
        v16h aNext = a;
        if (hasNext) aNext = loadAfrag(aRow + kb + 32, khalf);

#pragma unroll
        for (int j = 0; j < 8; ++j)
            acc[j] = __builtin_amdgcn_wmma_f32_16x16x32_f16(
                false, a, false, bf[j], (short)0, acc[j], false, false);

        __syncthreads();          // all reads done before next overwrite
        a = aNext;
        buf ^= 1;
    }

    // C/D layout: lane l, vgpr r -> M = m0 + wave*16 + (l>>4)*8 + r, N = n0+16j+(l&15)
    const int mBase = m0 + wave * 16 + khalf * 8;
#pragma unroll
    for (int j = 0; j < 8; ++j) {
        const int col = n0 + j * 16 + row;
        float bv = (ACT == ACT_BIAS_RELU) ? bias[col] : 0.0f;
#pragma unroll
        for (int r = 0; r < 8; ++r) {
            float v = acc[j][r];
            if (ACT == ACT_TANH)           v = tanhf(v);
            else if (ACT == ACT_BIAS_RELU) v = fmaxf(v + bv, 0.0f);
            Out[(size_t)(mBase + r) * ldo + col_off + col] = (OT)v;
        }
    }
}

// ---------------------------------------------------------------------------
// Per-node attention + aggregation. One wave per node n:
//   w[k]  = (h1[n*16+k,:] . u[n,:]) * scale (+ mask penalty)
//   softmax over k, agg[n,:] = sum_k w[k] * feat[n*16+k,:]   (feat f32, agg f16)
// ---------------------------------------------------------------------------
template <int DF>
__global__ __launch_bounds__(256) void attn_agg(
    const _Float16* __restrict__ h1,   // [N*16, 512] f16
    const float* __restrict__ u,       // [N, 512] f32
    const float* __restrict__ feat,    // [N*16, DF] f32 (raw features)
    const int* __restrict__ mask,      // may be nullptr
    _Float16* __restrict__ agg,        // [N, DF] f16
    float scale)
{
    const int lane = threadIdx.x & 31;
    const int n = blockIdx.x * 8 + (threadIdx.x >> 5);

    float uv[16];
    {
        const v4f* up = (const v4f*)(u + (size_t)n * 512 + lane * 16);
#pragma unroll
        for (int i = 0; i < 4; ++i) {
            v4f t = up[i];
            uv[4 * i] = t[0]; uv[4 * i + 1] = t[1]; uv[4 * i + 2] = t[2]; uv[4 * i + 3] = t[3];
        }
    }

    float w[16];
#pragma unroll
    for (int k = 0; k < 16; ++k) {
        v16h hv = *(const v16h*)(h1 + ((size_t)n * 16 + k) * 512 + lane * 16);
        float p = 0.0f;
#pragma unroll
        for (int i = 0; i < 16; ++i) p += (float)hv[i] * uv[i];
#pragma unroll
        for (int off = 16; off > 0; off >>= 1) p += __shfl_xor(p, off, 32);
        float s = p * scale;
        if (mask) s += -9999999.0f * (float)mask[(size_t)n * 16 + k];
        w[k] = s;
    }

    float mx = w[0];
#pragma unroll
    for (int k = 1; k < 16; ++k) mx = fmaxf(mx, w[k]);
    float sum = 0.0f;
#pragma unroll
    for (int k = 0; k < 16; ++k) { w[k] = __expf(w[k] - mx); sum += w[k]; }
    const float inv = 1.0f / sum;

    constexpr int PER = DF / 32;
    float av[PER];
#pragma unroll
    for (int i = 0; i < PER; ++i) av[i] = 0.0f;
#pragma unroll
    for (int k = 0; k < 16; ++k) {
        const float* fp = feat + ((size_t)n * 16 + k) * DF + lane * PER;
        const float wk = w[k] * inv;
#pragma unroll
        for (int i = 0; i < PER; ++i) av[i] += wk * fp[i];
    }
    _Float16* op = agg + (size_t)n * DF + lane * PER;
#pragma unroll
    for (int i = 0; i < PER; ++i) op[i] = (_Float16)av[i];
}

// ---------------------------------------------------------------------------
// One-pass f32 -> f16 cast (8 elements/thread, b128 in / b128 out)
// ---------------------------------------------------------------------------
__global__ __launch_bounds__(256) void cast_f16(const float* __restrict__ in,
                                                _Float16* __restrict__ out, int n)
{
    int i = (blockIdx.x * 256 + threadIdx.x) * 8;
    if (i >= n) return;
    const v4f* q = (const v4f*)(in + i);
    v4f a = q[0], b = q[1];
    v8h r;
#pragma unroll
    for (int j = 0; j < 4; ++j) { r[j] = (_Float16)a[j]; r[4 + j] = (_Float16)b[j]; }
    *(v8h*)(out + i) = r;
}

// 512x512 transpose + cast to f16 (weights only; tiny)
__global__ __launch_bounds__(256) void transpose_cast512(const float* __restrict__ in,
                                                         _Float16* __restrict__ out)
{
    int c = blockIdx.x * 16 + (threadIdx.x & 15);
    int r = blockIdx.y * 16 + (threadIdx.x >> 4);
    out[(size_t)c * 512 + r] = (_Float16)in[(size_t)r * 512 + c];
}

// ---------------------------------------------------------------------------
// Host launcher
// ---------------------------------------------------------------------------
extern "C" void kernel_launch(void* const* d_in, const int* in_sizes, int n_in,
                              void* d_out, int out_size, void* d_ws, size_t ws_size,
                              hipStream_t stream) {
    (void)in_sizes; (void)n_in; (void)out_size; (void)ws_size;

    const float* x     = (const float*)d_in[0];
    const float* neibs = (const float*)d_in[1];
    const float* edge  = (const float*)d_in[2];
    const int*   mask  = (const int*)d_in[3];
    const float* W1x = (const float*)d_in[4];
    const float* W2x = (const float*)d_in[5];
    const float* W1n = (const float*)d_in[6];
    const float* W2n = (const float*)d_in[7];
    const float* W1e = (const float*)d_in[8];
    const float* W2e = (const float*)d_in[9];
    const float* Wfx = (const float*)d_in[10];
    const float* bfx = (const float*)d_in[11];
    const float* Wfn = (const float*)d_in[12];
    const float* bfn = (const float*)d_in[13];
    const float* Wfe = (const float*)d_in[14];
    const float* bfe = (const float*)d_in[15];
    float* out = (float*)d_out;

    const int Nn = 8192, Kk = 16, D = 256, E = 128, H = 512, O = 256;
    const int NK = Nn * Kk;

    char* ws = (char*)d_ws;
    size_t off = 0;
    auto carve = [&](size_t bytes) -> char* {
        char* p = ws + off;
        off = (off + bytes + 255) & ~(size_t)255;
        return p;
    };

    _Float16* bufBig  = (_Float16*)carve((size_t)NK * H * 2);   // h1n then h1e
    _Float16* neibs16 = (_Float16*)carve((size_t)NK * D * 2);   // 64 MB (L2-resident)
    _Float16* edge16  = (_Float16*)carve((size_t)NK * E * 2);   // 32 MB
    _Float16* x16     = (_Float16*)carve((size_t)Nn * D * 2);
    _Float16* h1x     = (_Float16*)carve((size_t)Nn * H * 2);
    _Float16* xatt    = (_Float16*)carve((size_t)Nn * H * 2);
    float* u_n = (float*)carve((size_t)Nn * H * 4);
    float* u_e = (float*)carve((size_t)Nn * H * 4);
    _Float16* w1x16  = (_Float16*)carve((size_t)H * D * 2);
    _Float16* w2x16  = (_Float16*)carve((size_t)H * H * 2);
    _Float16* w1n16  = (_Float16*)carve((size_t)H * D * 2);
    _Float16* w1e16  = (_Float16*)carve((size_t)H * E * 2);
    _Float16* w2nT16 = (_Float16*)carve((size_t)H * H * 2);
    _Float16* w2eT16 = (_Float16*)carve((size_t)H * H * 2);
    _Float16* wfx16  = (_Float16*)carve((size_t)O * D * 2);
    _Float16* wfn16  = (_Float16*)carve((size_t)O * D * 2);
    _Float16* wfe16  = (_Float16*)carve((size_t)O * E * 2);
    _Float16* aggN16 = (_Float16*)carve((size_t)Nn * D * 2);
    _Float16* aggE16 = (_Float16*)carve((size_t)Nn * E * 2);

    dim3 tb(256);
    auto castBlocks = [](int n) { return (n / 8 + 255) / 256; };
    auto tilesFor   = [](int M, int N) { return (M >> 7) * (N >> 7); };
    const float invSqrt512 = 0.04419417382415922f;

    // ---- one-pass f32->f16 casts (weights + activations) ----
    cast_f16<<<castBlocks(Nn * D), tb, 0, stream>>>(x,     x16,     Nn * D);
    cast_f16<<<castBlocks(NK * D), tb, 0, stream>>>(neibs, neibs16, NK * D);
    cast_f16<<<castBlocks(NK * E), tb, 0, stream>>>(edge,  edge16,  NK * E);
    cast_f16<<<castBlocks(H * D),  tb, 0, stream>>>(W1x, w1x16, H * D);
    cast_f16<<<castBlocks(H * H),  tb, 0, stream>>>(W2x, w2x16, H * H);
    cast_f16<<<castBlocks(H * D),  tb, 0, stream>>>(W1n, w1n16, H * D);
    cast_f16<<<castBlocks(H * E),  tb, 0, stream>>>(W1e, w1e16, H * E);
    cast_f16<<<castBlocks(O * D),  tb, 0, stream>>>(Wfx, wfx16, O * D);
    cast_f16<<<castBlocks(O * D),  tb, 0, stream>>>(Wfn, wfn16, O * D);
    cast_f16<<<castBlocks(O * E),  tb, 0, stream>>>(Wfe, wfe16, O * E);
    transpose_cast512<<<dim3(32, 32), tb, 0, stream>>>(W2n, w2nT16);
    transpose_cast512<<<dim3(32, 32), tb, 0, stream>>>(W2e, w2eT16);

    // ---- x branch: h1x = tanh(x @ W1x.T); x_att = h1x @ W2x.T ----
    gemm_wmma<_Float16, ACT_TANH><<<tilesFor(Nn, H), tb, 0, stream>>>(
        x16, D, w1x16, D, h1x, H, 0, nullptr, Nn, H, D);
    gemm_wmma<_Float16, ACT_NONE><<<tilesFor(Nn, H), tb, 0, stream>>>(
        h1x, H, w2x16, H, xatt, H, 0, nullptr, Nn, H, H);
    // u_n = x_att @ W2n,  u_e = x_att @ W2e  (score GEMMs folded onto x_att)
    gemm_wmma<float, ACT_NONE><<<tilesFor(Nn, H), tb, 0, stream>>>(
        xatt, H, w2nT16, H, u_n, H, 0, nullptr, Nn, H, H);
    gemm_wmma<float, ACT_NONE><<<tilesFor(Nn, H), tb, 0, stream>>>(
        xatt, H, w2eT16, H, u_e, H, 0, nullptr, Nn, H, H);

    // ---- neighbor phase ----
    gemm_wmma<_Float16, ACT_TANH><<<tilesFor(NK, H), tb, 0, stream>>>(
        neibs16, D, w1n16, D, bufBig, H, 0, nullptr, NK, H, D);
    attn_agg<256><<<Nn / 8, tb, 0, stream>>>(bufBig, u_n, neibs, nullptr, aggN16, invSqrt512);

    // ---- edge phase (reuse bufBig) ----
    gemm_wmma<_Float16, ACT_TANH><<<tilesFor(NK, H), tb, 0, stream>>>(
        edge16, E, w1e16, E, bufBig, H, 0, nullptr, NK, H, E);
    attn_agg<128><<<Nn / 8, tb, 0, stream>>>(bufBig, u_e, edge, mask, aggE16, 1.0f);

    // ---- final projections, fused bias+ReLU, into [8192,768] f32 out ----
    gemm_wmma<float, ACT_BIAS_RELU><<<tilesFor(Nn, O), tb, 0, stream>>>(
        x16, D, wfx16, D, out, 768, 0, bfx, Nn, O, D);
    gemm_wmma<float, ACT_BIAS_RELU><<<tilesFor(Nn, O), tb, 0, stream>>>(
        aggN16, D, wfn16, D, out, 768, 256, bfn, Nn, O, D);
    gemm_wmma<float, ACT_BIAS_RELU><<<tilesFor(Nn, O), tb, 0, stream>>>(
        aggE16, E, wfe16, E, out, 768, 512, bfe, Nn, O, E);
}
